// CausalAttention_23819888623955
// MI455X (gfx1250) — compile-verified, matrix operand order checked by reading, not run on hardware
//
#include <hip/hip_runtime.h>
#include <hip/hip_bf16.h>
#include <math.h>

// ---------------------------------------------------------------------------
// CDNA5 (gfx1250, wave32) causal MHA forward.
// bf16 WMMA everywhere, async global->LDS double-buffered GEMM,
// software-pipelined flash attention (scalarized control flow, no spills).
// ---------------------------------------------------------------------------

typedef __bf16 bf16_t;
typedef __attribute__((ext_vector_type(16))) __bf16 v16bf;
typedef __attribute__((ext_vector_type(8)))  __bf16 v8bf;
typedef __attribute__((ext_vector_type(8)))  float  v8f;
typedef __attribute__((ext_vector_type(4)))  int    v4i;

__device__ __forceinline__ v8f wmma_bf16(v16bf a, v16bf b, v8f c) {
  // D = A(16x32) * B(32x16) + C, f32 accumulate
  return __builtin_amdgcn_wmma_f32_16x16x32_bf16(false, a, false, b, (short)0, c,
                                                 false, false);
}

// Load a 16x32 bf16 fragment in the CDNA5 WMMA 16-bit A/B per-lane pattern:
// lane L (r = L&15, h = L>>4) holds  [0..7]  = (row r, k = h*8+j)
//                                    [8..15] = (row r, k = 16+h*8+j)
// Works for global or LDS (generic) pointers; both halves are contiguous
// 16-byte chunks -> global_load_b128 / ds_read_b128.
__device__ __forceinline__ v16bf frag_ld(const bf16_t* __restrict__ p, int ld) {
  int lane = threadIdx.x & 31;
  int r = lane & 15, h = lane >> 4;
  const v8bf* lo = (const v8bf*)(p + (size_t)r * ld + h * 8);
  const v8bf* hi = (const v8bf*)(p + (size_t)r * ld + 16 + h * 8);
  v16bf o;
#pragma unroll
  for (int j = 0; j < 8; ++j) { o[j] = (*lo)[j]; o[j + 8] = (*hi)[j]; }
  return o;
}

// ---------------------------------------------------------------------------
// Async global -> LDS (ASYNCcnt path). The gfx1250 builtin takes typed
// addrspace(1)/addrspace(3) int4 pointers; build them via integer round-trip
// (low 32 bits of a generic shared pointer == hardware LDS offset).
// ---------------------------------------------------------------------------
#if __has_builtin(__builtin_amdgcn_global_load_async_to_lds_b128)
typedef __attribute__((address_space(1))) v4i gv4i_t;
typedef __attribute__((address_space(3))) v4i lv4i_t;
#endif

__device__ __forceinline__ void async_copy_b128(const void* gsrc, unsigned lds_off) {
#if __has_builtin(__builtin_amdgcn_global_load_async_to_lds_b128)
  __builtin_amdgcn_global_load_async_to_lds_b128(
      (gv4i_t*)(__UINTPTR_TYPE__)gsrc, (lv4i_t*)(__UINTPTR_TYPE__)lds_off, 0, 0);
#else
  asm volatile("global_load_async_to_lds_b128 %0, %1, off"
               :: "v"(lds_off), "v"((unsigned long long)(__UINTPTR_TYPE__)gsrc)
               : "memory");
#endif
}

__device__ __forceinline__ void wait_async0() {
#if __has_builtin(__builtin_amdgcn_s_wait_asynccnt)
  __builtin_amdgcn_s_wait_asynccnt(0);
#else
  asm volatile("s_wait_asynccnt 0x0" ::: "memory");
#endif
}

// ---------------------------------------------------------------------------
// f32 -> bf16 conversion
// ---------------------------------------------------------------------------
__global__ void cvt_f32_bf16(const float* __restrict__ in, bf16_t* __restrict__ out,
                             int n) {
  int i = blockIdx.x * blockDim.x + threadIdx.x;
  if (i < n) out[i] = (bf16_t)in[i];
}

// ---------------------------------------------------------------------------
// C[M,N] = A[M,K] @ W[N,K]^T (torch Linear), bf16 in, f32 accumulate.
// Block (256 thr, 8 waves) computes 128x64; wave (wm,wn) does 32x32 (2x2 WMMA).
// K-loop staged through double-buffered LDS tiles filled with async copies:
// copies for step k+32 overlap WMMAs on step k.
// LDS rows padded to 40 elems (80 B) so fragment ds_read_b128s tile the banks.
// ---------------------------------------------------------------------------
#define LDSPITCH 40

template <bool OUT_BF16, bool ADD_BIAS>
__global__ __launch_bounds__(256, 1)
void gemm_xWT(const bf16_t* __restrict__ A, const bf16_t* __restrict__ W,
              void* __restrict__ Cout, const float* __restrict__ bias,
              int M, int N, int Kd) {
  __shared__ __align__(16) bf16_t As[2][128][LDSPITCH];
  __shared__ __align__(16) bf16_t Ws[2][64][LDSPITCH];

  int nb = N >> 6;                     // blocks along N (64-wide)
  int bm = blockIdx.x / nb;
  int bn = blockIdx.x - bm * nb;
  int m0 = bm << 7, n0 = bn << 6;

  int tid = threadIdx.x;
  int w = tid >> 5;
  int wm = w >> 1, wn = w & 1;         // 4 x 2 waves
  int lane = tid & 31;
  int lr = lane & 15, lh = lane >> 4;

  const bf16_t* A0 = A + (size_t)m0 * Kd;   // 128 x K strip
  const bf16_t* W0 = W + (size_t)n0 * Kd;   //  64 x K strip

  // cooperative async fill of one K-step tile pair (128x32 A, 64x32 W)
  auto fill = [&](int buf, int k) {
    // A: 512 b128 chunks, 2 per thread
#pragma unroll
    for (int c = tid; c < 512; c += 256) {
      int row = c >> 2, ch = c & 3;
      async_copy_b128(A0 + (size_t)row * Kd + k + ch * 8,
                      (unsigned)(__UINTPTR_TYPE__)&As[buf][row][ch * 8]);
    }
    // W: 256 b128 chunks, 1 per thread
    {
      int row = tid >> 2, ch = tid & 3;
      async_copy_b128(W0 + (size_t)row * Kd + k + ch * 8,
                      (unsigned)(__UINTPTR_TYPE__)&Ws[buf][row][ch * 8]);
    }
  };

  fill(0, 0);

  v8f acc[2][2] = {{{}, {}}, {{}, {}}};
  for (int k = 0; k < Kd; k += 32) {
    int buf = (k >> 5) & 1;
    wait_async0();                     // this wave's copies into buf are done
    __syncthreads();                   // everyone's copies visible
    if (k + 32 < Kd) fill(buf ^ 1, k + 32);   // overlap next fill with compute

    v16bf a0 = frag_ld(&As[buf][wm * 32][0], LDSPITCH);
    v16bf a1 = frag_ld(&As[buf][wm * 32 + 16][0], LDSPITCH);
    v16bf b0 = frag_ld(&Ws[buf][wn * 32][0], LDSPITCH);
    v16bf b1 = frag_ld(&Ws[buf][wn * 32 + 16][0], LDSPITCH);
    acc[0][0] = wmma_bf16(a0, b0, acc[0][0]);
    acc[0][1] = wmma_bf16(a0, b1, acc[0][1]);
    acc[1][0] = wmma_bf16(a1, b0, acc[1][0]);
    acc[1][1] = wmma_bf16(a1, b1, acc[1][1]);
    __syncthreads();                   // done reading buf before it is refilled
  }

  // D layout: lane L holds rows i + 8*(L>>4), col = L&15
#pragma unroll
  for (int mi = 0; mi < 2; ++mi)
#pragma unroll
    for (int ni = 0; ni < 2; ++ni)
#pragma unroll
      for (int i = 0; i < 8; ++i) {
        int r = m0 + wm * 32 + mi * 16 + i + 8 * lh;
        int c = n0 + wn * 32 + ni * 16 + lr;
        float v = acc[mi][ni][i];
        if (ADD_BIAS) v += bias[c];
        if (OUT_BF16) ((bf16_t*)Cout)[(size_t)r * N + c] = (bf16_t)v;
        else          ((float*)Cout)[(size_t)r * N + c] = v;
      }
}

// ---------------------------------------------------------------------------
// Scatter QKV [B*T, 3E] -> Q[b,h,t,d], K[b,h,t,d], V^T[b,h,d,t]
// ---------------------------------------------------------------------------
__global__ void scatter_qkv(const bf16_t* __restrict__ qkv, bf16_t* __restrict__ Qb,
                            bf16_t* __restrict__ Kb, bf16_t* __restrict__ VTb,
                            int B, int T, int E, int H, int dk) {
  int idx = blockIdx.x * blockDim.x + threadIdx.x;
  int total = B * T * 3 * E;
  if (idx >= total) return;
  int c = idx % (3 * E);
  int r = idx / (3 * E);
  int b = r / T, t = r - b * T;
  int which = c / E;
  int hc = c - which * E;
  int h = hc / dk, d = hc - h * dk;
  bf16_t v = qkv[idx];
  size_t bh = (size_t)b * H + h;
  if (which == 0)      Qb[(bh * T + t) * dk + d] = v;
  else if (which == 1) Kb[(bh * T + t) * dk + d] = v;
  else                 VTb[(bh * dk + d) * T + t] = v;
}

// ---------------------------------------------------------------------------
// Causal flash attention: one wave per (b,h, 16-query tile).
// Computes S^T = K @ Q^T so the WMMA D layout (lane=query, elems=keys)
// repacks in-lane into the A-fragment of P for P@V (no cross-lane transpose).
// Wave id is readfirstlane'd so all control flow is scalar (EXEC stays all-1s,
// as WMMA requires) and the next K chunk is prefetched branchlessly with a
// clamped base.
// ---------------------------------------------------------------------------
__global__ __launch_bounds__(256, 1)
void attn_fwd(const bf16_t* __restrict__ Q, const bf16_t* __restrict__ K,
              const bf16_t* __restrict__ VT, bf16_t* __restrict__ O, int T) {
  const int dk = 64;
  // wave-uniform work id, made explicitly scalar for the compiler
  int wid = __builtin_amdgcn_readfirstlane(blockIdx.x * (blockDim.x >> 5) +
                                           (threadIdx.x >> 5));
  int qtiles = T >> 4;
  int bh = wid / qtiles;
  int qt = wid - bh * qtiles;
  int lane = threadIdx.x & 31;
  int lr = lane & 15, lh = lane >> 4;

  const bf16_t* Qb = Q + (size_t)bh * T * dk;
  const bf16_t* Kb = K + (size_t)bh * T * dk;
  const bf16_t* Vb = VT + (size_t)bh * dk * T;   // [dk][T]
  bf16_t*       Ob = O + (size_t)bh * T * dk;

  // Q^T B-fragments (lane = query column, elems along dk), held across k-loop
  v16bf qf0 = frag_ld(Qb + (size_t)(qt * 16) * dk + 0, dk);
  v16bf qf1 = frag_ld(Qb + (size_t)(qt * 16) * dk + 32, dk);

  v8f o0 = {}, o1 = {}, o2 = {}, o3 = {};
  float m = -1e30f, l = 0.f;
  const float scale = 0.125f;      // 1/sqrt(64)
  const float NEG = -1e30f;
  int qg = qt * 16 + lr;           // this lane's query index (per-lane)
  int qhi = qt * 16 + 15;          // scalar

  // prologue: K fragments for chunk 0
  v16bf k00 = frag_ld(Kb, dk);
  v16bf k10 = frag_ld(Kb + (size_t)16 * dk, dk);
  v16bf k01 = frag_ld(Kb + 32, dk);
  v16bf k11 = frag_ld(Kb + (size_t)16 * dk + 32, dk);

  for (int kb = 0; kb <= qhi; kb += 32) {
    // S^T tiles: s0 = keys kb..kb+15, s1 = keys kb+16..kb+31 (x 16 queries)
    v8f s0 = {}, s1 = {};
    s0 = wmma_bf16(k00, qf0, s0);
    s1 = wmma_bf16(k10, qf0, s1);
    s0 = wmma_bf16(k01, qf1, s0);
    s1 = wmma_bf16(k11, qf1, s1);

    // branchless prefetch of next chunk's K fragments (clamped base: one
    // harmless redundant prefetch of chunk 0 on the final iteration)
    int kn = kb + 32;
    int kc = (kn <= qhi) ? kn : 0;
    k00 = frag_ld(Kb + (size_t)kc * dk, dk);
    k10 = frag_ld(Kb + (size_t)(kc + 16) * dk, dk);
    k01 = frag_ld(Kb + (size_t)kc * dk + 32, dk);
    k11 = frag_ld(Kb + (size_t)(kc + 16) * dk + 32, dk);

    bool need_mask = (kb + 31 > qt * 16);
    float rmax = NEG;
#pragma unroll
    for (int i = 0; i < 8; ++i) {
      float v0 = s0[i] * scale;    // element i -> key kb + 8*lh + i
      float v1 = s1[i] * scale;    // element i -> key kb + 16 + 8*lh + i
      if (need_mask) {
        if (kb + 8 * lh + i > qg)      v0 = NEG;
        if (kb + 16 + 8 * lh + i > qg) v1 = NEG;
      }
      s0[i] = v0; s1[i] = v1;      // reuse accumulator registers
      rmax = fmaxf(rmax, fmaxf(v0, v1));
    }
    rmax = fmaxf(rmax, __shfl_xor(rmax, 16, 32));   // combine halves per query
    float mnew = fmaxf(m, rmax);
    float alpha = __expf(m - mnew);
    float rsum = 0.f;
    v16bf pf;                       // A-fragment of P (16 q x 32 keys), in-lane
#pragma unroll
    for (int i = 0; i < 8; ++i) {
      float e0 = __expf(s0[i] - mnew);
      float e1 = __expf(s1[i] - mnew);
      rsum += e0 + e1;
      pf[i] = (bf16_t)e0;
      pf[i + 8] = (bf16_t)e1;
    }
    rsum += __shfl_xor(rsum, 16, 32);
    l = l * alpha + rsum;
    m = mnew;
    // broadcast per-query rescale into the O accumulator row layout
    float af[8];
#pragma unroll
    for (int i = 0; i < 8; ++i) af[i] = __shfl(alpha, i + 8 * lh, 32);

    {
      v16bf vf = frag_ld(Vb + (size_t)0 * 16 * T + kb, T);
#pragma unroll
      for (int i = 0; i < 8; ++i) o0[i] *= af[i];
      o0 = wmma_bf16(pf, vf, o0);
    }
    {
      v16bf vf = frag_ld(Vb + (size_t)1 * 16 * T + kb, T);
#pragma unroll
      for (int i = 0; i < 8; ++i) o1[i] *= af[i];
      o1 = wmma_bf16(pf, vf, o1);
    }
    {
      v16bf vf = frag_ld(Vb + (size_t)2 * 16 * T + kb, T);
#pragma unroll
      for (int i = 0; i < 8; ++i) o2[i] *= af[i];
      o2 = wmma_bf16(pf, vf, o2);
    }
    {
      v16bf vf = frag_ld(Vb + (size_t)3 * 16 * T + kb, T);
#pragma unroll
      for (int i = 0; i < 8; ++i) o3[i] *= af[i];
      o3 = wmma_bf16(pf, vf, o3);
    }
  }

  float linv[8];
#pragma unroll
  for (int i = 0; i < 8; ++i) {
    float li = __shfl(l, i + 8 * lh, 32);
    linv[i] = 1.0f / li;
  }
  // O written in [b,h,t,d] memory order == the reference's buggy view(B,T,E)
#pragma unroll
  for (int i = 0; i < 8; ++i) {
    int r = qt * 16 + i + 8 * lh;
    Ob[(size_t)r * dk + 0 + lr]  = (bf16_t)(o0[i] * linv[i]);
    Ob[(size_t)r * dk + 16 + lr] = (bf16_t)(o1[i] * linv[i]);
    Ob[(size_t)r * dk + 32 + lr] = (bf16_t)(o2[i] * linv[i]);
    Ob[(size_t)r * dk + 48 + lr] = (bf16_t)(o3[i] * linv[i]);
  }
}

// ---------------------------------------------------------------------------
// Launch
// ---------------------------------------------------------------------------
extern "C" void kernel_launch(void* const* d_in, const int* in_sizes, int n_in,
                              void* d_out, int out_size, void* d_ws, size_t ws_size,
                              hipStream_t stream) {
  const float* x      = (const float*)d_in[0];
  const float* w_qkv  = (const float*)d_in[1];
  const float* w_proj = (const float*)d_in[2];
  const float* b_proj = (const float*)d_in[3];

  const int B = 2, T = 2048, E = 1024, H = 16, dk = 64;
  const int M = B * T;  // 4096

  size_t off = 0;
  auto carve = [&](size_t elems) -> bf16_t* {
    bf16_t* p = (bf16_t*)((char*)d_ws + off);
    off += ((elems * sizeof(bf16_t)) + 255) & ~(size_t)255;
    return p;
  };
  bf16_t* XB     = carve((size_t)M * E);          // x bf16
  bf16_t* WQKVB  = carve((size_t)3 * E * E);      // w_qkv bf16
  bf16_t* WPROJB = carve((size_t)E * E);          // w_proj bf16
  bf16_t* QKVB   = carve((size_t)M * 3 * E);      // qkv result
  bf16_t* QB     = carve((size_t)B * H * T * dk); // [b,h,t,d]
  bf16_t* KB     = carve((size_t)B * H * T * dk); // [b,h,t,d]
  bf16_t* VTB    = carve((size_t)B * H * dk * T); // [b,h,d,t]
  bf16_t* ATTNB  = carve((size_t)M * E);          // attention out (== view(B,T,E))
  (void)ws_size; (void)in_sizes; (void)n_in; (void)out_size;

  cvt_f32_bf16<<<(M * E) / 256, 256, 0, stream>>>(x, XB, M * E);
  cvt_f32_bf16<<<(3 * E * E) / 256, 256, 0, stream>>>(w_qkv, WQKVB, 3 * E * E);
  cvt_f32_bf16<<<(E * E) / 256, 256, 0, stream>>>(w_proj, WPROJB, E * E);

  {  // QKV projection: [4096,1024] @ [3072,1024]^T -> [4096,3072] bf16
    int blocks = (M / 128) * ((3 * E) / 64);       // 32 * 48 = 1536
    gemm_xWT<true, false><<<blocks, 256, 0, stream>>>(XB, WQKVB, (void*)QKVB,
                                                      nullptr, M, 3 * E, E);
  }
  {
    int total = M * 3 * E;
    scatter_qkv<<<total / 256, 256, 0, stream>>>(QKVB, QB, KB, VTB, B, T, E, H, dk);
  }
  {  // causal attention: 32 (b,h) x 128 q-tiles = 4096 waves
    int waves = B * H * (T / 16);
    attn_fwd<<<waves / 8, 256, 0, stream>>>(QB, KB, VTB, ATTNB, T);
  }
  {  // output projection + bias: [4096,1024] @ [1024,1024]^T -> f32 d_out
    int blocks = (M / 128) * (E / 64);             // 32 * 16 = 512
    gemm_xWT<false, true><<<blocks, 256, 0, stream>>>(ATTNB, WPROJB, d_out,
                                                      b_proj, M, E, E);
  }
}
